// MultiHeadAttention_78520592105487
// MI455X (gfx1250) — matrix operand order repaired and checked by reading, not compiled, not run on hardware
//
#include <hip/hip_runtime.h>
#include <hip/hip_bf16.h>

// ---------------------------------------------------------------------------
// MHA forward for MI455X (gfx1250, wave32, WMMA).
//   B=2, S=2048, D=1024, H=16, hd=64
// Stage 1: Y = X @ W^T + b in f16 WMMA (f32 accum), emitted head-split.
//          Q,K -> [b,h,s,64] ; V -> transposed [b,h,64,s] (for PV A-operand).
// Stage 2: klen[b] = popcount(key_padding_mask[b,:])  (prefix-valid mask).
// Stage 3: flash attention, online softmax, transposed-output PV, residual.
// ---------------------------------------------------------------------------

#define BATCH   2
#define S_LEN   2048
#define DMODEL  1024
#define NHEAD   16
#define HDIM    64

typedef _Float16 v16h __attribute__((ext_vector_type(16)));
typedef _Float16 v8h  __attribute__((ext_vector_type(8)));
typedef float    v8f  __attribute__((ext_vector_type(8)));

// ---- WMMA helper: D = A(16x32 f16) * B(32x16 f16) + C(16x16 f32) ----------
__device__ __forceinline__ v8f wmma_f16(v16h a, v16h b, v8f c) {
  return __builtin_amdgcn_wmma_f32_16x16x32_f16(
      /*neg_a=*/false, a, /*neg_b=*/false, b,
      /*c_mod=*/(short)0, c, /*reuse_a=*/false, /*reuse_b=*/false);
}

// A-fragment gather: halves j=0..15 at p0[0..15]
// (contiguous 16-bit A 16x32 layout: K = (lane>=16?16:0) + j).
__device__ __forceinline__ v16h ld_afrag(const _Float16* p0) {
  v8h lo = *(const v8h*)p0;
  v8h hi = *(const v8h*)(p0 + 8);
  v16h r;
#pragma unroll
  for (int i = 0; i < 8; ++i) { r[i] = lo[i]; r[i + 8] = hi[i]; }
  return r;
}

// ---------------------------------------------------------------------------
// Stage 1: projection GEMM.  Y[m,n] = sum_k X[m,k]*W[n,k] + bias[n]
// Block: 256 thr (8 waves), tile 128(M) x 128(N), K-step 32.
// mode 0 -> out[((b*H+h)*S+s)*64+d] ; mode 1 -> out[((b*H+h)*64+d)*S+s]
// ---------------------------------------------------------------------------
__global__ __launch_bounds__(256) void proj_kernel(
    const float* __restrict__ X, const float* __restrict__ W,
    const float* __restrict__ bias, _Float16* __restrict__ Y, int mode) {
  __shared__ _Float16 xs[128][32];
  __shared__ _Float16 wsh[128][32];

  const int tid  = threadIdx.x;
  const int lane = tid & 31;
  const int col  = lane & 15;
  const int half = lane >> 4;
  const int wv   = tid >> 5;
  const int wm   = wv & 3;   // 4 wave rows * 32
  const int wn   = wv >> 2;  // 2 wave cols * 64
  const int m0   = blockIdx.x * 128;
  const int n0   = blockIdx.y * 128;

  const int lrow = tid >> 1;
  const int lcol = (tid & 1) * 16;

  v8f acc[2][4];
#pragma unroll
  for (int i = 0; i < 2; ++i)
#pragma unroll
    for (int j = 0; j < 4; ++j) acc[i][j] = (v8f){};

  for (int kk = 0; kk < DMODEL; kk += 32) {
    const float4* xp = (const float4*)(X + (size_t)(m0 + lrow) * DMODEL + kk + lcol);
    const float4* wp = (const float4*)(W + (size_t)(n0 + lrow) * DMODEL + kk + lcol);
#pragma unroll
    for (int i = 0; i < 4; ++i) {
      float4 xv = xp[i];
      xs[lrow][lcol + i * 4 + 0] = (_Float16)xv.x;
      xs[lrow][lcol + i * 4 + 1] = (_Float16)xv.y;
      xs[lrow][lcol + i * 4 + 2] = (_Float16)xv.z;
      xs[lrow][lcol + i * 4 + 3] = (_Float16)xv.w;
      float4 wvv = wp[i];
      wsh[lrow][lcol + i * 4 + 0] = (_Float16)wvv.x;
      wsh[lrow][lcol + i * 4 + 1] = (_Float16)wvv.y;
      wsh[lrow][lcol + i * 4 + 2] = (_Float16)wvv.z;
      wsh[lrow][lcol + i * 4 + 3] = (_Float16)wvv.w;
    }
    __syncthreads();

    v16h a[2], b[4];
#pragma unroll
    for (int ab = 0; ab < 2; ++ab)
      a[ab] = ld_afrag(&xs[wm * 32 + ab * 16 + col][half * 16]);
#pragma unroll
    for (int bb = 0; bb < 4; ++bb)
      b[bb] = *(const v16h*)&wsh[wn * 64 + bb * 16 + col][half * 16];
#pragma unroll
    for (int ab = 0; ab < 2; ++ab)
#pragma unroll
      for (int bb = 0; bb < 4; ++bb)
        acc[ab][bb] = wmma_f16(a[ab], b[bb], acc[ab][bb]);
    __syncthreads();
  }

  // Epilogue: bias + convert + head-split (or transposed) store.
#pragma unroll
  for (int bb = 0; bb < 4; ++bb) {
    const int n  = n0 + wn * 64 + bb * 16 + col;
    const float bv = bias[n];
    const int hI = n / HDIM, dI = n % HDIM;
#pragma unroll
    for (int ab = 0; ab < 2; ++ab)
#pragma unroll
      for (int r = 0; r < 8; ++r) {
        const int m  = m0 + wm * 32 + ab * 16 + half * 8 + r;
        const int bI = m / S_LEN, sI = m % S_LEN;
        const float vv = acc[ab][bb][r] + bv;
        size_t off;
        if (mode == 0)
          off = (((size_t)(bI * NHEAD + hI) * S_LEN + sI) * HDIM + dI);
        else
          off = (((size_t)(bI * NHEAD + hI) * HDIM + dI) * S_LEN + sI);
        Y[off] = (_Float16)vv;
      }
  }
}

// ---------------------------------------------------------------------------
// Stage 2: klen[b] = number of valid keys (mask is a valid prefix).
// ---------------------------------------------------------------------------
__global__ __launch_bounds__(256) void klen_kernel(
    const unsigned char* __restrict__ kmask, int* __restrict__ klen) {
  __shared__ int cnt;
  if (threadIdx.x == 0) cnt = 0;
  __syncthreads();
  const int b = blockIdx.x;
  int local = 0;
  for (int i = threadIdx.x; i < S_LEN; i += blockDim.x)
    local += kmask[(size_t)b * S_LEN + i] ? 1 : 0;
  atomicAdd(&cnt, local);
  __syncthreads();
  if (threadIdx.x == 0) klen[b] = cnt;
}

// ---------------------------------------------------------------------------
// Stage 3: flash attention (causal + key padding), transposed-output PV.
// Grid: (S/64 q-tiles, B*H). Block: 128 thr = 4 waves, 16 q-rows per wave.
// ---------------------------------------------------------------------------
__global__ __launch_bounds__(128) void flash_kernel(
    const _Float16* __restrict__ Qh, const _Float16* __restrict__ Kh,
    const _Float16* __restrict__ Vt, const float* __restrict__ qin,
    const unsigned char* __restrict__ qmask, const int* __restrict__ klenp,
    float* __restrict__ out) {
  __shared__ _Float16 lp[4][16][32];   // per-wave P tile [q][kv]
  __shared__ float    lsc[4][16];      // per-wave per-row broadcast

  const int lane = threadIdx.x & 31;
  const int col  = lane & 15;
  const int half = lane >> 4;
  const int wv   = threadIdx.x >> 5;
  const int bh   = blockIdx.y;
  const int b    = bh / NHEAD;
  const int h    = bh % NHEAD;
  const int qbase = blockIdx.x * 64;
  const int qrow  = qbase + wv * 16 + col;  // A-frag row / output q per lane

  // Q A-fragments (d-steps 0..31, 32..63)
  const _Float16* qp = Qh + ((size_t)bh * S_LEN + qrow) * HDIM;
  v16h aq[2];
  aq[0] = ld_afrag(qp + half * 16);
  aq[1] = ld_afrag(qp + 32 + half * 16);

  const int klen = klenp[b];
  const int kend = min(klen, qbase + 64);
  const int nchunks = (kend + 31) >> 5;

  float mrow[8], lrow[8];
#pragma unroll
  for (int r = 0; r < 8; ++r) { mrow[r] = -1e30f; lrow[r] = 0.0f; }
  v8f oacc[4];  // O^T: 64(d) x 16(q); lane holds fixed q = col
#pragma unroll
  for (int f = 0; f < 4; ++f) oacc[f] = (v8f){};

  for (int c = 0; c < nchunks; ++c) {
    const int kb = c << 5;

    // K B-fragments: B(d x kv), element = K[kv][d]; contiguous in d.
    v16h bk[2][2];
#pragma unroll
    for (int nb = 0; nb < 2; ++nb) {
      const int kv = kb + nb * 16 + col;
      const _Float16* kp = Kh + ((size_t)bh * S_LEN + kv) * HDIM + half * 16;
      bk[nb][0] = *(const v16h*)(kp);
      bk[nb][1] = *(const v16h*)(kp + 32);
    }
    if (c + 1 < nchunks)  // prefetch next K chunk (global_prefetch_b8)
      __builtin_prefetch(Kh + ((size_t)bh * S_LEN + kb + 32 + lane) * HDIM, 0, 0);

    // S = Q * K^T  (16q x 32kv per wave)
    v8f sa[2] = {(v8f){}, (v8f){}};
#pragma unroll
    for (int ds = 0; ds < 2; ++ds) {
      sa[0] = wmma_f16(aq[ds], bk[0][ds], sa[0]);
      sa[1] = wmma_f16(aq[ds], bk[1][ds], sa[1]);
    }

    // scale + causal + key-padding masks (selects, EXEC stays all-1)
    float p[2][8], rmax[8];
#pragma unroll
    for (int r = 0; r < 8; ++r) {
      const int qr  = qbase + wv * 16 + half * 8 + r;
      const int kv0 = kb + col, kv1 = kb + 16 + col;
      float s0 = sa[0][r] * 0.125f;
      float s1 = sa[1][r] * 0.125f;
      s0 = ((kv0 <= qr) && (kv0 < klen)) ? s0 : -1e30f;
      s1 = ((kv1 <= qr) && (kv1 < klen)) ? s1 : -1e30f;
      p[0][r] = s0; p[1][r] = s1;
      rmax[r] = fmaxf(s0, s1);
    }
#pragma unroll
    for (int r = 0; r < 8; ++r)
#pragma unroll
      for (int m = 1; m <= 8; m <<= 1)
        rmax[r] = fmaxf(rmax[r], __shfl_xor(rmax[r], m, 32));

    float es[8], rsum[8];
#pragma unroll
    for (int r = 0; r < 8; ++r) {
      const float mn = fmaxf(mrow[r], rmax[r]);
      es[r]  = __expf(mrow[r] - mn);
      mrow[r] = mn;
      const float p0 = __expf(p[0][r] - mn);
      const float p1 = __expf(p[1][r] - mn);
      p[0][r] = p0; p[1][r] = p1;
      rsum[r] = p0 + p1;
    }
#pragma unroll
    for (int r = 0; r < 8; ++r)
#pragma unroll
      for (int m = 1; m <= 8; m <<= 1)
        rsum[r] += __shfl_xor(rsum[r], m, 32);
#pragma unroll
    for (int r = 0; r < 8; ++r) lrow[r] = lrow[r] * es[r] + rsum[r];

    if (col == 0)
#pragma unroll
      for (int r = 0; r < 8; ++r) lsc[wv][half * 8 + r] = es[r];
#pragma unroll
    for (int nb = 0; nb < 2; ++nb)
#pragma unroll
      for (int r = 0; r < 8; ++r)
        lp[wv][half * 8 + r][nb * 16 + col] = (_Float16)p[nb][r];
    __syncthreads();

    // rescale O^T accumulators: per-lane scalar (q = col)
    const float sc = lsc[wv][col];
#pragma unroll
    for (int f = 0; f < 4; ++f)
#pragma unroll
      for (int r = 0; r < 8; ++r) oacc[f][r] *= sc;

    // P^T B-fragment: B(kv x q) element = P[q][kv] -> contiguous lp row read.
    const v16h bp = *(const v16h*)&lp[wv][col][half * 16];

    // O^T += V^T * P^T ; A = V^T (d x kv), contiguous kv from Vt layout.
#pragma unroll
    for (int db = 0; db < 4; ++db) {
      const int d = db * 16 + col;
      const _Float16* vp = Vt + ((size_t)bh * HDIM + d) * S_LEN + kb + half * 16;
      const v16h av = ld_afrag(vp);
      oacc[db] = wmma_f16(av, bp, oacc[db]);
    }
    __syncthreads();
  }

  // Epilogue: 1/l, query-padding zeroing, residual add, store.
  if (col == 0)
#pragma unroll
    for (int r = 0; r < 8; ++r) lsc[wv][half * 8 + r] = lrow[r];
  __syncthreads();
  const float l = lsc[wv][col];
  const float scl =
      (qmask[(size_t)b * S_LEN + qrow] != 0) ? (1.0f / l) : 0.0f;
  const size_t obase = ((size_t)b * S_LEN + qrow) * DMODEL + h * HDIM;
#pragma unroll
  for (int db = 0; db < 4; ++db)
#pragma unroll
    for (int r = 0; r < 8; ++r) {
      const int d = db * 16 + half * 8 + r;
      out[obase + d] = oacc[db][r] * scl + qin[obase + d];
    }
}

// ---------------------------------------------------------------------------
extern "C" void kernel_launch(void* const* d_in, const int* in_sizes, int n_in,
                              void* d_out, int out_size, void* d_ws,
                              size_t ws_size, hipStream_t stream) {
  (void)in_sizes; (void)n_in; (void)out_size; (void)ws_size;
  const float* q  = (const float*)d_in[0];
  const float* k  = (const float*)d_in[1];
  const float* v  = (const float*)d_in[2];
  const float* Wq = (const float*)d_in[3];
  const float* bq = (const float*)d_in[4];
  const float* Wk = (const float*)d_in[5];
  const float* bk = (const float*)d_in[6];
  const float* Wv = (const float*)d_in[7];
  const float* bv = (const float*)d_in[8];
  const unsigned char* kmask = (const unsigned char*)d_in[9];
  const unsigned char* qmask = (const unsigned char*)d_in[10];
  float* out = (float*)d_out;

  const size_t QKV_BYTES = (size_t)BATCH * NHEAD * S_LEN * HDIM * sizeof(_Float16);
  char* ws = (char*)d_ws;
  _Float16* Qh  = (_Float16*)(ws);
  _Float16* Kh  = (_Float16*)(ws + QKV_BYTES);
  _Float16* Vt  = (_Float16*)(ws + 2 * QKV_BYTES);
  int*      kln = (int*)(ws + 3 * QKV_BYTES);

  dim3 pg((BATCH * S_LEN) / 128, DMODEL / 128);   // 32 x 8
  proj_kernel<<<pg, 256, 0, stream>>>(q, Wq, bq, Qh, 0);
  proj_kernel<<<pg, 256, 0, stream>>>(k, Wk, bk, Kh, 0);
  proj_kernel<<<pg, 256, 0, stream>>>(v, Wv, bv, Vt, 1);

  klen_kernel<<<BATCH, 256, 0, stream>>>(kmask, kln);

  dim3 fg(S_LEN / 64, BATCH * NHEAD);             // 32 x 32
  flash_kernel<<<fg, 128, 0, stream>>>(Qh, Kh, Vt, q, qmask, kln, out);
}
